// CRFAE_17875653886131
// MI455X (gfx1250) — matrix operand order repaired and checked by reading, not compiled, not run on hardware
//
#include <hip/hip_runtime.h>

typedef __attribute__((ext_vector_type(16))) _Float16 v16h;
typedef __attribute__((ext_vector_type(8)))  float    v8f;

#define Bsz 512
#define Ssz 256
#define Lsz 128
#define BTILE 16

// ---------------------------------------------------------------------------
// Kernel 0: precompute column maxes c_j = max_i T[i][j] and W = exp(T - c_j)
// (f16). W is constant across all 255 time steps, so amortized to zero cost.
// ---------------------------------------------------------------------------
__global__ __launch_bounds__(Lsz) void crfae_prep(const float* __restrict__ trans,
                                                  _Float16* __restrict__ Wexp,
                                                  float* __restrict__ cmax)
{
    const int j = threadIdx.x;           // 0..127, one block
    float m = -__builtin_inff();
    for (int i = 0; i < Lsz; ++i) m = fmaxf(m, trans[i * Lsz + j]);
    cmax[j] = m;
    for (int i = 0; i < Lsz; ++i)
        Wexp[i * Lsz + j] = (_Float16)__expf(trans[i * Lsz + j] - m);
}

// ---------------------------------------------------------------------------
// Kernel 1: CRF forward recurrence. One workgroup = 16 batches, 8 waves;
// wave n owns output columns [16n, 16n+16). Per step both alpha and beta
// updates are a 16x128 @ 128x128 GEMM done with v_wmma_f32_16x16x32_f16.
// Previous state kept in registers for a branchless mask select.
// ---------------------------------------------------------------------------
__global__ __launch_bounds__(256) void crfae_main(
    const int* __restrict__ words,
    const float* __restrict__ emits,
    const unsigned char* __restrict__ maskp,
    const float* __restrict__ ftab,
    const float* __restrict__ startv,
    const float* __restrict__ endv,
    const _Float16* __restrict__ Wexp,
    const float* __restrict__ cmax,
    float* __restrict__ diffs)
{
    __shared__ float    la[BTILE][Lsz + 4];
    __shared__ float    lb[BTILE][Lsz + 4];
    __shared__ _Float16 ua[BTILE][Lsz + 8];
    __shared__ _Float16 ub[BTILE][Lsz + 8];
    __shared__ float    amax[BTILE];
    __shared__ float    bmax[BTILE];
    __shared__ int      wrd[BTILE];
    __shared__ unsigned char msk[BTILE];

    const int tid  = threadIdx.x;
    const int b0   = blockIdx.x * BTILE;
    const int ln   = tid & 31;
    const int wv   = tid >> 5;           // wave id 0..7 -> N tile
    const int N0   = wv * 16;
    const int ncol = ln & 15;
    const int hs   = ln >> 4;            // half-wave select

    // Persistent B fragments of W for this wave's N tile (contiguous K order, same as A:
    // lanes 0-15 hold K=kc*32+0..15, lanes 16-31 hold K=kc*32+16..31).
    v16h bw[4];
    #pragma unroll
    for (int kc = 0; kc < 4; ++kc) {
        const int kbase = kc * 32 + hs * 16;
        v16h f;
        #pragma unroll
        for (int e = 0; e < 16; ++e)
            f[e] = Wexp[(kbase + e) * Lsz + (N0 + ncol)];
        bw[kc] = f;
    }
    const float cN = cmax[N0 + ncol];

    // init: la = start + emits[:,0]; lb = la + feature_table[words[:,0]]
    {
        const int b  = tid >> 4;
        const int jb = (tid & 15) * 8;
        const int w0 = words[(b0 + b) * Ssz + 0];
        #pragma unroll
        for (int e = 0; e < 8; ++e) {
            const int j = jb + e;
            const float a0 = startv[j] + emits[((size_t)(b0 + b) * Ssz + 0) * Lsz + j];
            la[b][j] = a0;
            lb[b][j] = a0 + ftab[(size_t)w0 * Lsz + j];
        }
    }
    __syncthreads();

    // Previous state for this thread's phase-3 tile elements, in registers:
    // element r <-> (row m = r + 8*hs, col N0+ncol).
    float prevA[8], prevB[8];
    #pragma unroll
    for (int r = 0; r < 8; ++r) {
        const int m = r + hs * 8;
        prevA[r] = la[m][N0 + ncol];
        prevB[r] = lb[m][N0 + ncol];
    }

    for (int t = 1; t < Ssz; ++t) {
        // ---- phase 1: row maxes, u = exp(state - max) f16, stage words/mask ----
        {
            const int b  = tid >> 4;
            const int jb = (tid & 15) * 8;
            if (tid < BTILE) {
                wrd[tid] = words[(b0 + tid) * Ssz + t];
                msk[tid] = maskp[(b0 + tid) * Ssz + t];
            }
            float va[8], vb[8];
            float ma = -__builtin_inff(), mb = -__builtin_inff();
            #pragma unroll
            for (int e = 0; e < 8; ++e) {
                va[e] = la[b][jb + e];
                vb[e] = lb[b][jb + e];
                ma = fmaxf(ma, va[e]);
                mb = fmaxf(mb, vb[e]);
            }
            #pragma unroll
            for (int off = 1; off < 16; off <<= 1) {
                ma = fmaxf(ma, __shfl_xor(ma, off, 32));
                mb = fmaxf(mb, __shfl_xor(mb, off, 32));
            }
            if ((tid & 15) == 0) { amax[b] = ma; bmax[b] = mb; }
            #pragma unroll
            for (int e = 0; e < 8; ++e) {
                ua[b][jb + e] = (_Float16)__expf(va[e] - ma);
                ub[b][jb + e] = (_Float16)__expf(vb[e] - mb);
            }
        }
        __syncthreads();

        // ---- phase 2: s = u @ W  (8x v_wmma_f32_16x16x32_f16, K=128) ----
        v8f sa = {};
        v8f sb = {};
        {
            const int arow = ln & 15;    // A layout: lanes 0-15 / 16-31 both M=0..15
            #pragma unroll
            for (int kc = 0; kc < 4; ++kc) {
                const int kb = kc * 32 + hs * 16; // halves 0-15: K=kb..kb+15
                v16h aa, ab;
                #pragma unroll
                for (int e = 0; e < 8; ++e) {
                    aa[e]     = ua[arow][kb + e];
                    aa[8 + e] = ua[arow][kb + 8 + e];
                    ab[e]     = ub[arow][kb + e];
                    ab[8 + e] = ub[arow][kb + 8 + e];
                }
                sa = __builtin_amdgcn_wmma_f32_16x16x32_f16(false, aa, false, bw[kc],
                                                            (short)0, sa, false, false);
                sb = __builtin_amdgcn_wmma_f32_16x16x32_f16(false, ab, false, bw[kc],
                                                            (short)0, sb, false, false);
            }
        }

        // ---- phase 3: branchless state update for this wave's 16x16 tile ----
        // D layout: VGPR r, lane ln -> row M = r + 8*hs, col N = N0 + ln%16.
        #pragma unroll
        for (int r = 0; r < 8; ++r) {
            const int m  = r + hs * 8;
            const int gb = b0 + m;
            const int j  = N0 + ncol;
            const float ev = emits[((size_t)gb * Ssz + t) * Lsz + j];
            const float dv = ftab[(size_t)wrd[m] * Lsz + j];
            const bool  mk = msk[m] != 0;
            const float av = ev + amax[m] + cN + __logf(sa[r]);
            const float bv = ev + dv + bmax[m] + cN + __logf(sb[r]);
            prevA[r] = mk ? av : prevA[r];
            prevB[r] = mk ? bv : prevB[r];
        }
        #pragma unroll
        for (int r = 0; r < 8; ++r) {
            const int m = r + hs * 8;
            la[m][N0 + ncol] = prevA[r];
            lb[m][N0 + ncol] = prevB[r];
        }
        __syncthreads();
    }

    // ---- final: logsumexp_j(state + end) per batch row, write diff ----
    {
        const int b  = tid >> 4;
        const int jb = (tid & 15) * 8;
        float va[8], vb[8];
        float ma = -__builtin_inff(), mb = -__builtin_inff();
        #pragma unroll
        for (int e = 0; e < 8; ++e) {
            va[e] = la[b][jb + e] + endv[jb + e];
            vb[e] = lb[b][jb + e] + endv[jb + e];
            ma = fmaxf(ma, va[e]);
            mb = fmaxf(mb, vb[e]);
        }
        #pragma unroll
        for (int off = 1; off < 16; off <<= 1) {
            ma = fmaxf(ma, __shfl_xor(ma, off, 32));
            mb = fmaxf(mb, __shfl_xor(mb, off, 32));
        }
        float ssa = 0.f, ssb = 0.f;
        #pragma unroll
        for (int e = 0; e < 8; ++e) {
            ssa += __expf(va[e] - ma);
            ssb += __expf(vb[e] - mb);
        }
        #pragma unroll
        for (int off = 1; off < 16; off <<= 1) {
            ssa += __shfl_xor(ssa, off, 32);
            ssb += __shfl_xor(ssb, off, 32);
        }
        if ((tid & 15) == 0)
            diffs[b0 + b] = (ma + __logf(ssa)) - (mb + __logf(ssb));
    }
}

// ---------------------------------------------------------------------------
// Kernel 2: deterministic fixed-order reduction of the 512 per-batch diffs.
// ---------------------------------------------------------------------------
__global__ void crfae_reduce(const float* __restrict__ diffs, float* __restrict__ out)
{
    if (threadIdx.x == 0 && blockIdx.x == 0) {
        float s = 0.f;
        for (int i = 0; i < Bsz; ++i) s += diffs[i];
        out[0] = s;
    }
}

extern "C" void kernel_launch(void* const* d_in, const int* in_sizes, int n_in,
                              void* d_out, int out_size, void* d_ws, size_t ws_size,
                              hipStream_t stream) {
    (void)in_sizes; (void)n_in; (void)out_size; (void)ws_size;

    const int*           words  = (const int*)d_in[0];
    const float*         emits  = (const float*)d_in[1];
    const unsigned char* maskp  = (const unsigned char*)d_in[2];
    const float*         ftab   = (const float*)d_in[3];
    const float*         startv = (const float*)d_in[4];
    const float*         trans  = (const float*)d_in[5];
    const float*         endv   = (const float*)d_in[6];
    float*               out    = (float*)d_out;

    char* ws = (char*)d_ws;
    _Float16* Wexp  = (_Float16*)ws;                                  // 32 KB
    float*    cmaxp = (float*)(ws + (size_t)Lsz * Lsz * 2);           // 512 B
    float*    diffs = (float*)(ws + (size_t)Lsz * Lsz * 2 + Lsz * 4); // 2 KB

    crfae_prep<<<1, Lsz, 0, stream>>>(trans, Wexp, cmaxp);
    crfae_main<<<Bsz / BTILE, 256, 0, stream>>>(words, emits, maskp, ftab,
                                                startv, endv, Wexp, cmaxp, diffs);
    crfae_reduce<<<1, 1, 0, stream>>>(diffs, out);
}
